// MultiHeadedAttention_14293651161768
// MI455X (gfx1250) — compile-verified, matrix operand order checked by reading, not run on hardware
//
#include <hip/hip_runtime.h>

// ---------------------------------------------------------------------------
// MI455X (gfx1250) causal MHA forward.
// f16 WMMA (v_wmma_f32_16x16x32_f16), fp32 accumulation, flash-attention
// streaming softmax, TDM (tensor_load_to_lds) double-buffered LDS staging.
// B=2, S=2048, D=1024, H=16, HD=64.
// ---------------------------------------------------------------------------

typedef __attribute__((ext_vector_type(16))) _Float16 v16h;
typedef __attribute__((ext_vector_type(8)))  _Float16 v8h;
typedef __attribute__((ext_vector_type(8)))  float    v8f;
typedef __attribute__((ext_vector_type(4)))  unsigned int v4u;
typedef __attribute__((ext_vector_type(8)))  int      v8i;
typedef __attribute__((ext_vector_type(4)))  int      v4i;

#define WMMA_F16(a, b, c) \
  __builtin_amdgcn_wmma_f32_16x16x32_f16(false, (a), false, (b), (short)0, (c), false, false)

// Generic pointer to LDS: low 32 bits are the wave-relative LDS byte address
// (flat aperture maps LDS via addr[31:0], ISA 10.2).
static __device__ inline unsigned lds_addr32(const void* p) {
  return (unsigned)(unsigned long long)p;
}

// Tensor Data Mover: load a 2D f16 tile (tile_d1 rows x tile_d0 elements,
// global row stride = stride0 elements) into LDS, adding pad_amount DWORDs
// of LDS padding every 2^(pad_interval+1) DWORDs (ISA 8.3/8.4 D# layout).
// tensor dims == tile dims -> no OOB clamping ever triggers.
static __device__ inline void tdm_load_2d_f16(unsigned lds_byte_addr,
                                              const void* gsrc,
                                              unsigned tile_d0, unsigned tile_d1,
                                              unsigned long long stride0,
                                              unsigned pad_interval_code,
                                              unsigned pad_amount_code) {
  unsigned long long ga = (unsigned long long)gsrc;
  v4u g0;
  g0[0] = 1u;                                                  // count=1, user D#
  g0[1] = lds_byte_addr;                                       // lds_addr
  g0[2] = (unsigned)ga;                                        // global_addr lo
  g0[3] = ((unsigned)(ga >> 32) & 0x01FFFFFFu) | (2u << 30);   // addr hi | type=2
  v8i g1;
  g1[0] = (int)((1u << 16)                                     // data_size = 2B
              | (1u << 20)                                     // pad_enable
              | (pad_interval_code << 22)
              | (pad_amount_code << 25));
  g1[1] = (int)(tile_d0 << 16);          // tensor_dim0[15:0]  @ bits 63:48
  g1[2] = (int)(tile_d1 << 16);          // dim0 hi=0 | tensor_dim1[15:0]
  g1[3] = (int)(tile_d0 << 16);          // dim1 hi=0 | tile_dim0
  g1[4] = (int)(tile_d1);                // tile_dim1 | tile_dim2=0
  g1[5] = (int)(unsigned)stride0;        // tensor_dim0_stride lo
  g1[6] = (int)((unsigned)(stride0 >> 32) & 0xFFFFu);  // stride hi | dim1_stride=0
  g1[7] = 0;
  v4i z4 = {};
#if defined(__clang_major__) && (__clang_major__ >= 23)
  v8i z8 = {};
  __builtin_amdgcn_tensor_load_to_lds(g0, g1, z4, z4, z8, 0);
#else
  __builtin_amdgcn_tensor_load_to_lds(g0, g1, z4, z4, 0);
#endif
}

// Load a 16x32 f16 A-fragment (or B-fragment of a column-major-stored matrix)
// per ISA 7.12.2: lane&15 = row/col, halves 0..7 -> K = 8*(lane>>4)+i,
// halves 8..15 -> +16.  Two 16-byte contiguous loads (ds/global _b128).
static __device__ inline v16h load_frag(const _Float16* base, int ld) {
  const int lane = threadIdx.x & 31;
  const int rc   = lane & 15;
  const int kg   = (lane >> 4) << 3;
  const _Float16* p = base + rc * ld + kg;
  v8h lo = *(const v8h*)(p);
  v8h hi = *(const v8h*)(p + 16);
  v16h r;
#pragma unroll
  for (int i = 0; i < 8; ++i) { r[i] = lo[i]; r[i + 8] = hi[i]; }
  return r;
}

// ------------------------------ convert ------------------------------------
__global__ void cvt_f32_f16(const float* __restrict__ in,
                            _Float16* __restrict__ out, int n) {
  int i = blockIdx.x * 256 + threadIdx.x;
  if (i < n) out[i] = (_Float16)in[i];
}

// ------------------------------ GEMM ---------------------------------------
// C[M,N] = A[M,K] * B[K,N], f16 in / fp32 accum.
// 128 threads = 4 waves; block tile 64x64; wave tile 32x32 (4 accumulators).
// A tile staged by TDM, B tile by manual transposed stores; both double-
// buffered with next-tile prefetch overlapping the WMMA work.
// MODE 0: fp32 C row-major.  MODE 1: scatter into head-major Q/K/V (f16).
template <int MODE>
__global__ void __launch_bounds__(128)
gemm_f16(const _Float16* __restrict__ A, const _Float16* __restrict__ Bm,
         int M, int N, int K,
         float* __restrict__ Cout,
         _Float16* __restrict__ Qb, _Float16* __restrict__ Kb,
         _Float16* __restrict__ Vb) {
  const int LDA = 40;                        // 32+8 halves, 80B rows
  __shared__ _Float16 As[2][64 * 40];        // [buf][row][k]
  __shared__ _Float16 Bs[2][64 * 40];        // [buf][col][k]  (B^T)
  const int m0 = blockIdx.x * 64;
  const int n0 = blockIdx.y * 64;
  const int t = threadIdx.x;
  const int wave = t >> 5, lane = t & 31;
  const int rw = (wave >> 1) * 32;
  const int cw = (wave & 1) * 32;

  auto issueA = [&](int k0, int buf) {       // wave0 only; 64 rows x 32 halves
    tdm_load_2d_f16(lds_addr32(&As[buf][0]), A + (size_t)m0 * K + k0,
                    /*tile_d0=*/32, /*tile_d1=*/64, (unsigned long long)K,
                    /*interval 16DW=*/3u, /*pad 4DW=*/3u);
  };
  auto loadB = [&](int k0, int buf) {        // all threads; transposed store
    int kk = t & 31, j0 = (t >> 5) * 16;
    const _Float16* src = Bm + (size_t)(k0 + kk) * N + n0 + j0;
    v8h s0 = *(const v8h*)src;
    v8h s1 = *(const v8h*)(src + 8);
#pragma unroll
    for (int i = 0; i < 8; ++i) {
      Bs[buf][(j0 + i) * LDA + kk]     = s0[i];
      Bs[buf][(j0 + 8 + i) * LDA + kk] = s1[i];
    }
  };

  v8f acc00 = {}, acc01 = {}, acc10 = {}, acc11 = {};

  if (wave == 0) issueA(0, 0);
  loadB(0, 0);
  if (wave == 0) __builtin_amdgcn_s_wait_tensorcnt(0);
  __syncthreads();

  for (int k0 = 0; k0 < K; k0 += 32) {
    const int cur = (k0 >> 5) & 1;
    if (k0 + 32 < K) {                       // prefetch next tile
      if (wave == 0) issueA(k0 + 32, cur ^ 1);
      loadB(k0 + 32, cur ^ 1);
    }
    v16h a0 = load_frag(&As[cur][rw * LDA], LDA);
    v16h a1 = load_frag(&As[cur][(rw + 16) * LDA], LDA);
    v16h b0 = load_frag(&Bs[cur][cw * LDA], LDA);
    v16h b1 = load_frag(&Bs[cur][(cw + 16) * LDA], LDA);
    acc00 = WMMA_F16(a0, b0, acc00);
    acc01 = WMMA_F16(a0, b1, acc01);
    acc10 = WMMA_F16(a1, b0, acc10);
    acc11 = WMMA_F16(a1, b1, acc11);
    if (wave == 0) __builtin_amdgcn_s_wait_tensorcnt(0);
    __syncthreads();
  }

  // Epilogue.  C/D layout: row M = r + 8*(lane>>4), col N = lane&15.
  const int lg = lane >> 4, ln = lane & 15;
#pragma unroll
  for (int q = 0; q < 4; ++q) {
    v8f acc = (q == 0) ? acc00 : (q == 1) ? acc01 : (q == 2) ? acc10 : acc11;
    int tm = rw + ((q >> 1) * 16);
    int tn = cw + ((q & 1) * 16);
    int col = n0 + tn + ln;
#pragma unroll
    for (int r = 0; r < 8; ++r) {
      int row = m0 + tm + lg * 8 + r;
      float v = acc[r];
      if (MODE == 0) {
        Cout[(size_t)row * N + col] = v;
      } else {
        // col in [0,3072): 0..1023 -> Q, 1024..2047 -> K, 2048..3071 -> V
        int which = col >> 10;
        int d = col & 1023;
        int h = d >> 6, hd = d & 63;
        int b = row >> 11;                   // / 2048
        int s = row & 2047;
        _Float16 hv = (_Float16)v;
        size_t bh = (size_t)(b * 16 + h);
        if (which == 0)      Qb[(bh * 2048 + s) * 64 + hd] = hv;   // [b,h,s,hd]
        else if (which == 1) Kb[(bh * 2048 + s) * 64 + hd] = hv;   // [b,h,s,hd]
        else                 Vb[(bh * 64 + hd) * 2048 + s] = hv;   // [b,h,hd,s]
      }
    }
  }
}

// --------------------------- flash attention -------------------------------
// 128 threads = 4 waves; block = (b,h, 64 query rows); wave = 16 query rows.
// K / V^T tiles streamed by TDM into double-buffered LDS with next-tile
// prefetch; online softmax; fp32 O accumulators.
__global__ void __launch_bounds__(128)
attn_flash(const _Float16* __restrict__ Qg, const _Float16* __restrict__ Kg,
           const _Float16* __restrict__ Vg, _Float16* __restrict__ Yg) {
  const int LDK = 72, LDV = 40, LDP = 40;
  __shared__ _Float16 Ks[2][32 * 72];      // [buf][key][hd]
  __shared__ _Float16 Vt[2][64 * 40];      // [buf][hd][key]
  __shared__ _Float16 Ps[4 * 16 * 40];     // per-wave P staging (C -> A layout)

  const int t = threadIdx.x, wave = t >> 5, lane = t & 31;
  const int lg = lane >> 4, ln = lane & 15;
  const int qt = blockIdx.x & 31, bh = blockIdx.x >> 5;
  const int q0 = qt * 64;
  const int q0w = q0 + wave * 16;
  const int b = bh >> 4, h = bh & 15;

  const _Float16* Qp = Qg + (size_t)bh * 2048 * 64;
  const _Float16* Kp = Kg + (size_t)bh * 2048 * 64;
  const _Float16* Vp = Vg + (size_t)bh * 64 * 2048;

  auto issueKV = [&](int kt, int buf) {    // wave0 only
    const int key0 = kt << 5;
    // K tile: 32 keys x 64 hd, row stride 64 (contiguous), pad 128B->+16B
    tdm_load_2d_f16(lds_addr32(&Ks[buf][0]), Kp + (size_t)key0 * 64,
                    /*tile_d0=*/64, /*tile_d1=*/32, 64ull,
                    /*interval 32DW=*/4u, /*pad 4DW=*/3u);
    // V^T tile: 64 hd x 32 keys, row stride S=2048, pad 64B->+16B
    tdm_load_2d_f16(lds_addr32(&Vt[buf][0]), Vp + key0,
                    /*tile_d0=*/32, /*tile_d1=*/64, 2048ull,
                    /*interval 16DW=*/3u, /*pad 4DW=*/3u);
  };

  // Q fragments loaded once, straight from global (rows contiguous 128B).
  v16h qf0 = load_frag(Qp + (size_t)q0w * 64, 64);       // hd 0..31
  v16h qf1 = load_frag(Qp + (size_t)q0w * 64 + 32, 64);  // hd 32..63

  v8f o0 = {}, o1 = {}, o2 = {}, o3 = {};                // O: 16 x 64 fp32
  float m[8], l[8];
#pragma unroll
  for (int r = 0; r < 8; ++r) { m[r] = -1e30f; l[r] = 0.f; }

  const int ktiles = (q0 + 64) >> 5;       // causal: keys <= q0+63

  if (wave == 0) {
    issueKV(0, 0);
    __builtin_amdgcn_s_wait_tensorcnt(0);
  }
  __syncthreads();

  for (int kt = 0; kt < ktiles; ++kt) {
    const int key0 = kt << 5;
    const int cur = kt & 1;
    if (wave == 0 && kt + 1 < ktiles) issueKV(kt + 1, cur ^ 1);  // prefetch

    // S = Q @ K^T : 16 q-rows x 32 keys, contract over hd=64 (2 chunks)
    v8f sc0 = {}, sc1 = {};
    v16h kb00 = load_frag(&Ks[cur][0], LDK);                 // keys 0-15,  hd 0-31
    v16h kb01 = load_frag(&Ks[cur][32], LDK);                // keys 0-15,  hd 32-63
    v16h kb10 = load_frag(&Ks[cur][16 * LDK], LDK);          // keys 16-31, hd 0-31
    v16h kb11 = load_frag(&Ks[cur][16 * LDK + 32], LDK);     // keys 16-31, hd 32-63
    sc0 = WMMA_F16(qf0, kb00, sc0);
    sc0 = WMMA_F16(qf1, kb01, sc0);
    sc1 = WMMA_F16(qf0, kb10, sc1);
    sc1 = WMMA_F16(qf1, kb11, sc1);

    // Online softmax.  Row M lives across the 16 lanes of one half-wave.
    const float scale = 0.125f;            // 1/sqrt(64)
    float p0[8], p1[8], alpha[8];
#pragma unroll
    for (int r = 0; r < 8; ++r) {
      int qrow = q0w + lg * 8 + r;
      float s0 = (key0 + ln <= qrow)      ? sc0[r] * scale : -1e30f;
      float s1 = (key0 + 16 + ln <= qrow) ? sc1[r] * scale : -1e30f;
      float mx = fmaxf(s0, s1);
      mx = fmaxf(mx, __shfl_xor(mx, 1, 32));
      mx = fmaxf(mx, __shfl_xor(mx, 2, 32));
      mx = fmaxf(mx, __shfl_xor(mx, 4, 32));
      mx = fmaxf(mx, __shfl_xor(mx, 8, 32));
      float mn = fmaxf(m[r], mx);
      alpha[r] = __expf(m[r] - mn);
      m[r] = mn;
      p0[r] = __expf(s0 - mn);
      p1[r] = __expf(s1 - mn);
      float ps = p0[r] + p1[r];
      ps += __shfl_xor(ps, 1, 32);
      ps += __shfl_xor(ps, 2, 32);
      ps += __shfl_xor(ps, 4, 32);
      ps += __shfl_xor(ps, 8, 32);
      l[r] = l[r] * alpha[r] + ps;
    }
#pragma unroll
    for (int r = 0; r < 8; ++r) {
      o0[r] *= alpha[r]; o1[r] *= alpha[r];
      o2[r] *= alpha[r]; o3[r] *= alpha[r];
    }
    // P: C-layout -> LDS -> A-layout fragment for P @ V
    _Float16* Pw = Ps + wave * 16 * LDP;
#pragma unroll
    for (int r = 0; r < 8; ++r) {
      Pw[(lg * 8 + r) * LDP + ln]      = (_Float16)p0[r];
      Pw[(lg * 8 + r) * LDP + 16 + ln] = (_Float16)p1[r];
    }
    __syncthreads();                       // P visible; K reads done
    v16h pf  = load_frag(Pw, LDP);
    v16h vb0 = load_frag(&Vt[cur][0], LDV);             // hd 0-15
    v16h vb1 = load_frag(&Vt[cur][16 * LDV], LDV);      // hd 16-31
    v16h vb2 = load_frag(&Vt[cur][32 * LDV], LDV);      // hd 32-47
    v16h vb3 = load_frag(&Vt[cur][48 * LDV], LDV);      // hd 48-63
    o0 = WMMA_F16(pf, vb0, o0);
    o1 = WMMA_F16(pf, vb1, o1);
    o2 = WMMA_F16(pf, vb2, o2);
    o3 = WMMA_F16(pf, vb3, o3);

    if (wave == 0) __builtin_amdgcn_s_wait_tensorcnt(0);  // next tiles landed
    __syncthreads();
  }

  // y[b][q][h*64 + c*16 + ln] = O / l
#pragma unroll
  for (int r = 0; r < 8; ++r) {
    int qrow = q0w + lg * 8 + r;
    float inv = 1.0f / l[r];
    size_t base = ((size_t)b * 2048 + qrow) * 1024 + h * 64 + ln;
    Yg[base]      = (_Float16)(o0[r] * inv);
    Yg[base + 16] = (_Float16)(o1[r] * inv);
    Yg[base + 32] = (_Float16)(o2[r] * inv);
    Yg[base + 48] = (_Float16)(o3[r] * inv);
  }
}

// ------------------------------ launcher -----------------------------------
extern "C" void kernel_launch(void* const* d_in, const int* in_sizes, int n_in,
                              void* d_out, int out_size, void* d_ws, size_t ws_size,
                              hipStream_t stream) {
  const float* x      = (const float*)d_in[0];   // [2,2048,1024]
  const float* w_attn = (const float*)d_in[1];   // [1024,3072]
  const float* w_proj = (const float*)d_in[2];   // [1024,1024]
  float* out = (float*)d_out;                    // [2,2048,1024] fp32

  char* ws = (char*)d_ws;                        // layout (MB offsets):
  _Float16* x16  = (_Float16*)(ws);              //  0: x f16      (8MB)
  _Float16* wa16 = (_Float16*)(ws + (size_t)8  * 1024 * 1024);  //  8: w_attn (6MB)
  _Float16* wp16 = (_Float16*)(ws + (size_t)14 * 1024 * 1024);  // 14: w_proj (2MB)
  _Float16* Qb   = (_Float16*)(ws + (size_t)16 * 1024 * 1024);  // 16: Q      (8MB)
  _Float16* Kb   = (_Float16*)(ws + (size_t)24 * 1024 * 1024);  // 24: K      (8MB)
  _Float16* Vb   = (_Float16*)(ws + (size_t)32 * 1024 * 1024);  // 32: V^T    (8MB)
  _Float16* Yb   = (_Float16*)(ws + (size_t)40 * 1024 * 1024);  // 40: y f16  (8MB)

  const int nx = 4096 * 1024, na = 1024 * 3072, np = 1024 * 1024;
  cvt_f32_f16<<<(nx + 255) / 256, 256, 0, stream>>>(x, x16, nx);
  cvt_f32_f16<<<(na + 255) / 256, 256, 0, stream>>>(w_attn, wa16, na);
  cvt_f32_f16<<<(np + 255) / 256, 256, 0, stream>>>(w_proj, wp16, np);

  // QKV projection: [4096,1024] x [1024,3072] -> head-major Q/K/V (f16)
  dim3 g1(4096 / 64, 3072 / 64);
  gemm_f16<1><<<g1, 128, 0, stream>>>(x16, wa16, 4096, 3072, 1024,
                                      nullptr, Qb, Kb, Vb);

  // Flash attention: 2*16 (b,h) x 32 q-tiles of 64 rows
  attn_flash<<<32 * 32, 128, 0, stream>>>(Qb, Kb, Vb, Yb);

  // Output projection: [4096,1024] x [1024,1024] -> fp32 out
  dim3 g2(4096 / 64, 1024 / 64);
  gemm_f16<0><<<g2, 128, 0, stream>>>(Yb, wp16, 4096, 1024, 1024,
                                      out, nullptr, nullptr, nullptr);
}
